// SparseRNN_54863912239446
// MI455X (gfx1250) — matrix operand order repaired and checked by reading, not compiled, not run on hardware
//
#include <hip/hip_runtime.h>
#include <hip/hip_bf16.h>
#include <math.h>

#define IN    2048
#define HID   8192
#define BATCH 32
#define SEQL  64

typedef __attribute__((ext_vector_type(16))) _Float16 v16h;
typedef __attribute__((ext_vector_type(8)))  _Float16 v8h;
typedef __attribute__((ext_vector_type(8)))  float    v8f;
typedef __attribute__((ext_vector_type(4)))  float    v4f;

// ---------------------------------------------------------------------------
// Setup kernels (run once per launch; deterministic)
// ---------------------------------------------------------------------------

__global__ void cvt_x_f16(const float* __restrict__ x, _Float16* __restrict__ xh, int n) {
  int i = blockIdx.x * blockDim.x + threadIdx.x;
  int stride = gridDim.x * blockDim.x;
  for (; i < n; i += stride) xh[i] = (_Float16)x[i];
}

// f16 atomic add via 32-bit CAS (duplicate COO coords are rare but exist)
__device__ inline void atomic_add_f16(_Float16* addr, float val) {
  uintptr_t ai = (uintptr_t)addr;
  unsigned* wp = (unsigned*)(ai & ~(uintptr_t)3);
  const bool hi = (ai & 2) != 0;
  unsigned old = *wp, assumed;
  do {
    assumed = old;
    unsigned short bits = hi ? (unsigned short)(assumed >> 16)
                             : (unsigned short)(assumed & 0xffffu);
    _Float16 cur = __builtin_bit_cast(_Float16, bits);
    _Float16 nv  = (_Float16)((float)cur + val);
    unsigned short nb = __builtin_bit_cast(unsigned short, nv);
    unsigned next = hi ? ((assumed & 0x0000ffffu) | ((unsigned)nb << 16))
                       : ((assumed & 0xffff0000u) | (unsigned)nb);
    old = atomicCAS(wp, assumed, next);
  } while (old != assumed);
}

__global__ void scatter_coo_f16(const int* __restrict__ rows, const int* __restrict__ cols,
                                const float* __restrict__ vals, int nnz,
                                _Float16* __restrict__ W, int kdim) {
  int i = blockIdx.x * blockDim.x + threadIdx.x;
  if (i < nnz) {
    atomic_add_f16(&W[(size_t)rows[i] * kdim + cols[i]], vals[i]);
  }
}

// ---------------------------------------------------------------------------
// One RNN timestep: PRE = Wih@x_t + Whh@h + bias ; h' = tanh(PRE)
// Each wave computes a 16(M) x 32(N=batch) strip via two 16x16x32 f16 WMMA
// chains sharing the A fragment. Weights stay resident in the 192MB L2.
//
// Layouts used here (C/D per CDNA5 ISA §7.12.2; A/B in one contiguous k order):
//   A (16x32 f16): lanes 0-15 row M=lane, K 0..15 contiguous;
//                  lanes 16-31 row M=lane-16, K 16..31 contiguous
//   B (32x16 f16): lanes 0-15 col N=lane, K 0..15 contiguous;
//                  lanes 16-31 col N=lane-16, K 16..31 contiguous
//   C/D (16x16 f32): VGPR v, lane l -> M = v + (l>=16 ? 8 : 0), N = l&15
// ---------------------------------------------------------------------------

__global__ void __launch_bounds__(128)
rnn_step_wmma(const _Float16* __restrict__ Wih,   // [HID][IN]
              const _Float16* __restrict__ Whh,   // [HID][HID]
              const _Float16* __restrict__ xh,    // [BATCH][SEQL][IN]
              const _Float16* __restrict__ hprev, // [BATCH][HID]
              const float*    __restrict__ bias,  // [HID]
              _Float16*       __restrict__ hnew,  // [BATCH][HID]
              float*          __restrict__ out,   // [BATCH][SEQL][HID]
              int t)
{
  const int lane = threadIdx.x & 31;
  const int wave = threadIdx.x >> 5;
  const int m0   = (blockIdx.x * 4 + wave) * 16;   // 512 M-tiles total
  const int nlo  = lane & 15;
  const int kgrp = lane >> 4;

  v8f acc0, acc1;
#pragma unroll
  for (int v = 0; v < 8; ++v) {
    float bv = bias[m0 + kgrp * 8 + v];
    acc0[v] = bv;
    acc1[v] = bv;
  }

  // ---- W_ih @ x_t  (K = IN) ----
  {
    const _Float16* a0 = Wih + (size_t)(m0 + nlo) * IN + kgrp * 16;
    const _Float16* b0 = xh + (size_t)nlo        * (SEQL * IN) + (size_t)t * IN + kgrp * 16;
    const _Float16* b1 = xh + (size_t)(nlo + 16) * (SEQL * IN) + (size_t)t * IN + kgrp * 16;
#pragma unroll 2
    for (int k = 0; k < IN; k += 32) {
      v8h alo = *(const v8h*)(a0 + k);
      v8h ahi = *(const v8h*)(a0 + k + 8);
      v16h a  = __builtin_shufflevector(alo, ahi, 0,1,2,3,4,5,6,7,8,9,10,11,12,13,14,15);
      v16h bb0 = *(const v16h*)(b0 + k);
      v16h bb1 = *(const v16h*)(b1 + k);
      acc0 = __builtin_amdgcn_wmma_f32_16x16x32_f16(false, a, false, bb0, (short)0, acc0, false, false);
      acc1 = __builtin_amdgcn_wmma_f32_16x16x32_f16(false, a, false, bb1, (short)0, acc1, false, false);
    }
  }

  // ---- W_hh @ h  (K = HID) ----
  {
    const _Float16* a0 = Whh + (size_t)(m0 + nlo) * HID + kgrp * 16;
    const _Float16* b0 = hprev + (size_t)nlo        * HID + kgrp * 16;
    const _Float16* b1 = hprev + (size_t)(nlo + 16) * HID + kgrp * 16;
#pragma unroll 2
    for (int k = 0; k < HID; k += 32) {
      v8h alo = *(const v8h*)(a0 + k);
      v8h ahi = *(const v8h*)(a0 + k + 8);
      v16h a  = __builtin_shufflevector(alo, ahi, 0,1,2,3,4,5,6,7,8,9,10,11,12,13,14,15);
      v16h bb0 = *(const v16h*)(b0 + k);
      v16h bb1 = *(const v16h*)(b1 + k);
      acc0 = __builtin_amdgcn_wmma_f32_16x16x32_f16(false, a, false, bb0, (short)0, acc0, false, false);
      acc1 = __builtin_amdgcn_wmma_f32_16x16x32_f16(false, a, false, bb1, (short)0, acc1, false, false);
    }
  }

  // ---- epilogue: tanh, write fp32 output + f16 next-state ----
  {
    v4f o0lo, o0hi, o1lo, o1hi;
    v8h h0, h1;
#pragma unroll
    for (int v = 0; v < 8; ++v) {
      float y0 = tanhf(acc0[v]);
      float y1 = tanhf(acc1[v]);
      if (v < 4) { o0lo[v] = y0; o1lo[v] = y1; }
      else       { o0hi[v - 4] = y0; o1hi[v - 4] = y1; }
      h0[v] = (_Float16)y0;
      h1[v] = (_Float16)y1;
    }
    const size_t mo = (size_t)(m0 + kgrp * 8);
    float* op0 = out + (size_t)nlo        * (SEQL * HID) + (size_t)t * HID + mo;
    float* op1 = out + (size_t)(nlo + 16) * (SEQL * HID) + (size_t)t * HID + mo;
    *(v4f*)op0       = o0lo;
    *(v4f*)(op0 + 4) = o0hi;
    *(v4f*)op1       = o1lo;
    *(v4f*)(op1 + 4) = o1hi;
    *(v8h*)(hnew + (size_t)nlo        * HID + mo) = h0;
    *(v8h*)(hnew + (size_t)(nlo + 16) * HID + mo) = h1;
  }
}

// ---------------------------------------------------------------------------
// Fallback (if workspace is too small for dense f16 weights): atomic SpMM
// ---------------------------------------------------------------------------

__global__ void fb_init_pre(const float* __restrict__ bias, float* __restrict__ pre) {
  int i = blockIdx.x * blockDim.x + threadIdx.x;
  if (i < HID * BATCH) pre[i] = bias[i >> 5];
}

__global__ void fb_spmm_ih(const int* __restrict__ rows, const int* __restrict__ cols,
                           const float* __restrict__ vals, int nnz,
                           const float* __restrict__ x, int t, float* __restrict__ pre) {
  int k = blockIdx.x * (blockDim.x >> 5) + (threadIdx.x >> 5);
  int b = threadIdx.x & 31;
  if (k < nnz) {
    int r = rows[k], c = cols[k];
    float v = vals[k];
    atomicAdd(&pre[r * BATCH + b], v * x[(size_t)b * (SEQL * IN) + (size_t)t * IN + c]);
  }
}

__global__ void fb_spmm_hh(const int* __restrict__ rows, const int* __restrict__ cols,
                           const float* __restrict__ vals, int nnz,
                           const float* __restrict__ h, float* __restrict__ pre) {
  int k = blockIdx.x * (blockDim.x >> 5) + (threadIdx.x >> 5);
  int b = threadIdx.x & 31;
  if (k < nnz) {
    int r = rows[k], c = cols[k];
    float v = vals[k];
    atomicAdd(&pre[r * BATCH + b], v * h[(size_t)c * BATCH + b]);
  }
}

__global__ void fb_tanh(const float* __restrict__ pre, float* __restrict__ h,
                        float* __restrict__ out, int t) {
  int i = blockIdx.x * blockDim.x + threadIdx.x;
  if (i < HID * BATCH) {
    float y = tanhf(pre[i]);
    h[i] = y;
    int r = i >> 5, b = i & 31;
    out[(size_t)b * (SEQL * HID) + (size_t)t * HID + r] = y;
  }
}

// ---------------------------------------------------------------------------
// Host launcher
// ---------------------------------------------------------------------------

extern "C" void kernel_launch(void* const* d_in, const int* in_sizes, int n_in,
                              void* d_out, int out_size, void* d_ws, size_t ws_size,
                              hipStream_t stream) {
  (void)n_in; (void)out_size;
  const float* x      = (const float*)d_in[0];
  const int*   idx_ih = (const int*)d_in[1];
  const float* val_ih = (const float*)d_in[2];
  const int*   idx_hh = (const int*)d_in[3];
  const float* val_hh = (const float*)d_in[4];
  const float* bias   = (const float*)d_in[5];
  float* out = (float*)d_out;
  const int nnz_ih = in_sizes[2];
  const int nnz_hh = in_sizes[4];

  const size_t whh_b = (size_t)HID * HID * sizeof(_Float16);   // 128 MB
  const size_t wih_b = (size_t)HID * IN  * sizeof(_Float16);   //  32 MB
  const size_t xh_b  = (size_t)BATCH * SEQL * IN * sizeof(_Float16); // 8 MB
  const size_t h_b   = (size_t)BATCH * HID * sizeof(_Float16);       // 512 KB
  const size_t need  = whh_b + wih_b + xh_b + 2 * h_b;

  char* ws = (char*)d_ws;
  if (ws_size >= need) {
    _Float16* Whh = (_Float16*)ws;
    _Float16* Wih = (_Float16*)(ws + whh_b);
    _Float16* xh  = (_Float16*)(ws + whh_b + wih_b);
    _Float16* hA  = (_Float16*)(ws + whh_b + wih_b + xh_b);
    _Float16* hB  = (_Float16*)(ws + whh_b + wih_b + xh_b + h_b);

    hipMemsetAsync(ws, 0, whh_b + wih_b, stream);   // zero dense weights
    hipMemsetAsync((void*)hA, 0, 2 * h_b, stream);  // h0 = 0

    cvt_x_f16<<<1024, 256, 0, stream>>>(x, xh, BATCH * SEQL * IN);
    scatter_coo_f16<<<(nnz_ih + 255) / 256, 256, 0, stream>>>(
        idx_ih, idx_ih + nnz_ih, val_ih, nnz_ih, Wih, IN);
    scatter_coo_f16<<<(nnz_hh + 255) / 256, 256, 0, stream>>>(
        idx_hh, idx_hh + nnz_hh, val_hh, nnz_hh, Whh, HID);

    _Float16* hp = hA;
    _Float16* hn = hB;
    for (int t = 0; t < SEQL; ++t) {
      rnn_step_wmma<<<HID / 64, 128, 0, stream>>>(Wih, Whh, xh, hp, bias, hn, out, t);
      _Float16* tmp = hp; hp = hn; hn = tmp;
    }
  } else {
    const size_t fb_need = (size_t)HID * BATCH * sizeof(float) * 2;
    if (ws_size < fb_need) return;
    float* pre = (float*)ws;
    float* h   = pre + (size_t)HID * BATCH;
    hipMemsetAsync((void*)h, 0, (size_t)HID * BATCH * sizeof(float), stream);
    for (int t = 0; t < SEQL; ++t) {
      fb_init_pre<<<(HID * BATCH + 255) / 256, 256, 0, stream>>>(bias, pre);
      fb_spmm_ih<<<(nnz_ih + 7) / 8, 256, 0, stream>>>(idx_ih, idx_ih + nnz_ih,
                                                       val_ih, nnz_ih, x, t, pre);
      fb_spmm_hh<<<(nnz_hh + 7) / 8, 256, 0, stream>>>(idx_hh, idx_hh + nnz_hh,
                                                       val_hh, nnz_hh, h, pre);
      fb_tanh<<<(HID * BATCH + 255) / 256, 256, 0, stream>>>(pre, h, out, t);
    }
  }
}
